// SE3EquivariantNetwork_35966056137501
// MI455X (gfx1250) — matrix operand order repaired and checked by reading, not compiled, not run on hardware
//
#include <hip/hip_runtime.h>
#include <hip/hip_bf16.h>
#include <stdint.h>
#include <stddef.h>

#define B_   2
#define N_   2048
#define KNN  16
#define COLS (B_*3*N_)   // 12288  (col = (b*3+vec)*N + n)
#define COLSN (B_*N_)    // 4096

typedef __attribute__((ext_vector_type(16))) __bf16 v16bf;
typedef __attribute__((ext_vector_type(8)))  float  v8f;

union BfVec { v16bf v; unsigned int u[8]; };

__device__ inline unsigned int packbf2(float lo, float hi) {
  unsigned int a = __float_as_uint(lo);
  unsigned int b = __float_as_uint(hi);
  a = (a + 0x7FFFu + ((a >> 16) & 1u)) >> 16;   // round-to-nearest-even
  b = (b + 0x7FFFu + ((b >> 16) & 1u)) >> 16;
  return (a & 0xFFFFu) | (b << 16);
}

// Swizzled packed-bf16 activation layout (B-operand ready):
// for K-slab ks=k/32, col-tile ct=m/16: lane (m%16)+16*((k>>4)&1) owns 8
// contiguous u32, u32 j holds K = 32*ks + 16*half + {2j, 2j+1}.
__device__ inline size_t swz(int k, int m, int M) {
  int ks = k >> 5;
  int half = (k >> 4) & 1;
  return ((size_t)(ks * (M >> 4) + (m >> 4)) * 32 + (m & 15) + (half << 4)) * 8
         + ((k & 15) >> 1);
}

// ---------------------------------------------------------------------------
// Weight packer: Wp[o][i/2] = bf16pair(W[o][2i], W[o][2i+1])
// ---------------------------------------------------------------------------
__global__ void pack_weight_kernel(const float* __restrict__ W,
                                   unsigned int* __restrict__ Wp, long total)
{
  long tid = (long)blockIdx.x * blockDim.x + threadIdx.x;
  if (tid >= total) return;
  Wp[tid] = packbf2(W[2 * tid], W[2 * tid + 1]);
}

// ---------------------------------------------------------------------------
// WMMA GEMM: Y[O,M] = W[O,I] * X[I,M] (+Add), bf16 operands pre-packed.
// Wp: [O][I/2] u32 pairs. Xp: swizzled packed. Outputs: f32 linear (Yf)
// and/or swizzled packed (Yp). O%16==0, I%32==0, M%64==0.
// One wave -> 16 x 64 output strip; software-pipelined K loop (ping-pong).
// ---------------------------------------------------------------------------
__global__ __launch_bounds__(128) void wmma_gemm_kernel(
    const unsigned int* __restrict__ Wp, const unsigned int* __restrict__ Xp,
    const float* __restrict__ Add, float* __restrict__ Yf,
    unsigned int* __restrict__ Yp, int O, int I, int M)
{
  const int lane = threadIdx.x & 31;
  const int mTiles = M >> 6;
  const int gw = blockIdx.x * 4 + (threadIdx.x >> 5);
  if (gw >= (O >> 4) * mTiles) return;     // wave-uniform guard
  const int oT = gw / mTiles, mT = gw % mTiles;
  const int hi = lane >> 4;
  const int Ih = I >> 1;
  const int row = (oT << 4) + (lane & 15);
  const int nct = M >> 4;

  v8f acc[4];
  #pragma unroll
  for (int t = 0; t < 4; ++t)
    #pragma unroll
    for (int r = 0; r < 8; ++r) acc[t][r] = 0.0f;

  const unsigned int* aRun = Wp + (size_t)row * Ih + (hi << 3);
  const unsigned int* bRun = Xp + ((size_t)mT * 128 + lane) * 8;
  const size_t bStride = (size_t)M * 16;   // u32 per K-slab

  BfVec a0v, a1v, b0v[4], b1v[4];
  auto loadA = [&](BfVec& a, const unsigned int* p) {
    #pragma unroll
    for (int j = 0; j < 8; ++j) { a.u[j] = p[j]; }
  };
  auto loadB = [&](BfVec* b, const unsigned int* p) {
    #pragma unroll
    for (int t = 0; t < 4; ++t)
      #pragma unroll
      for (int j = 0; j < 8; ++j) b[t].u[j] = p[t * 256 + j];
  };
  auto mma4 = [&](const BfVec& a, const BfVec* b) {
    #pragma unroll
    for (int t = 0; t < 4; ++t)
      acc[t] = __builtin_amdgcn_wmma_f32_16x16x32_bf16(
          false, a.v, false, b[t].v, (short)0, acc[t], false, false);
  };

  const int steps = I >> 5;                // >= 1
  loadA(a0v, aRun); loadB(b0v, bRun);      // step 0
  int s = 1;
  for (;;) {
    if (s == steps) { mma4(a0v, b0v); break; }
    aRun += 16; bRun += bStride;
    loadA(a1v, aRun); loadB(b1v, bRun);    // prefetch step s
    mma4(a0v, b0v); ++s;                   // consume step s-1
    if (s == steps) { mma4(a1v, b1v); break; }
    aRun += 16; bRun += bStride;
    loadA(a0v, aRun); loadB(b0v, bRun);
    mma4(a1v, b1v); ++s;
  }

  const int oRow = (oT << 4) + (hi << 3);
  #pragma unroll
  for (int t = 0; t < 4; ++t) {
    const int col = (mT << 6) + (t << 4) + (lane & 15);
    float v[8];
    #pragma unroll
    for (int r = 0; r < 8; ++r) {
      v[r] = acc[t][r];
      if (Add) v[r] += Add[(size_t)(oRow + r) * M + col];
    }
    if (Yf) {
      #pragma unroll
      for (int r = 0; r < 8; ++r) Yf[(size_t)(oRow + r) * M + col] = v[r];
    }
    if (Yp) {   // k = oRow + 2rp -> ks = oT/2, half = oT&1, j = hi*4 + rp
      unsigned int* yp =
          Yp + ((size_t)((oT >> 1) * nct + (mT << 2) + t) * 32 + (lane & 15) +
                ((oT & 1) << 4)) * 8 + (hi << 2);
      #pragma unroll
      for (int rp = 0; rp < 4; ++rp)
        yp[rp] = packbf2(v[2 * rp], v[2 * rp + 1]);
    }
  }
}

// Naive f32 GEMM for skinny / tiny shapes
__global__ void gemm_naive_kernel(const float* __restrict__ W,
                                  const float* __restrict__ X,
                                  float* __restrict__ Y, int O, int I, int M)
{
  long tid = (long)blockIdx.x * blockDim.x + threadIdx.x;
  if (tid >= (long)O * M) return;
  int o = (int)(tid / M), m = (int)(tid % M);
  float s = 0.0f;
  for (int i = 0; i < I; ++i) s += W[(size_t)o * I + i] * X[(size_t)i * M + m];
  Y[(size_t)o * M + m] = s;
}

// Naive GEMM emitting f32 + packed (for W_pre, K=3): thread per (o-pair, m)
__global__ void gemm_naive_pack_kernel(const float* __restrict__ W,
                                       const float* __restrict__ X,
                                       float* __restrict__ Yf,
                                       unsigned int* __restrict__ Yp,
                                       int O, int I, int M)
{
  long tid = (long)blockIdx.x * blockDim.x + threadIdx.x;
  if (tid >= (long)(O / 2) * M) return;
  int op = (int)(tid / M), m = (int)(tid % M);
  int o0 = 2 * op;
  float s0 = 0.0f, s1 = 0.0f;
  for (int i = 0; i < I; ++i) {
    float xv = X[(size_t)i * M + m];
    s0 += W[(size_t)o0 * I + i] * xv;
    s1 += W[(size_t)(o0 + 1) * I + i] * xv;
  }
  Yf[(size_t)o0 * M + m] = s0;
  Yf[(size_t)(o0 + 1) * M + m] = s1;
  Yp[swz(o0, m, M)] = packbf2(s0, s1);
}

// ---------------------------------------------------------------------------
// VN leaky-relu on channel pairs; emits f32 (optional) + packed (optional)
// ---------------------------------------------------------------------------
__global__ void vn_lrelu_pack_kernel(const float* __restrict__ x,
                                     const float* __restrict__ d,
                                     float* __restrict__ outF,
                                     unsigned int* __restrict__ outP,
                                     int C2, int N, int Bv, float slope)
{
  long tid = (long)blockIdx.x * blockDim.x + threadIdx.x;
  if (tid >= (long)Bv * C2 * N) return;
  int n  = (int)(tid % N);
  int cc = (int)((tid / N) % C2);
  int b  = (int)(tid / ((long)N * C2));
  size_t B3N = (size_t)Bv * 3 * N;
  float o[2][3];
  #pragma unroll
  for (int p = 0; p < 2; ++p) {
    size_t base = (size_t)(2 * cc + p) * B3N + (size_t)b * 3 * N + n;
    float x0 = x[base], x1 = x[base + N], x2 = x[base + 2 * (size_t)N];
    float d0 = d[base], d1 = d[base + N], d2 = d[base + 2 * (size_t)N];
    float dot = x0*d0 + x1*d1 + x2*d2;
    float dsq = d0*d0 + d1*d1 + d2*d2;
    if (dot >= 0.0f) { o[p][0] = x0; o[p][1] = x1; o[p][2] = x2; }
    else {
      float s = dot / (dsq + 1e-6f), c1 = 1.0f - slope;
      o[p][0] = slope*x0 + c1*(x0 - s*d0);
      o[p][1] = slope*x1 + c1*(x1 - s*d1);
      o[p][2] = slope*x2 + c1*(x2 - s*d2);
    }
    if (outF) {
      outF[base] = o[p][0]; outF[base + N] = o[p][1];
      outF[base + 2 * (size_t)N] = o[p][2];
    }
  }
  if (outP) {
    #pragma unroll
    for (int comp = 0; comp < 3; ++comp) {
      int m = b * 3 * N + comp * N + n;
      outP[swz(2 * cc, m, (int)B3N)] = packbf2(o[0][comp], o[1][comp]);
    }
  }
}

// ---------------------------------------------------------------------------
// Reductions (LDS tree)
// ---------------------------------------------------------------------------
__device__ inline float block_reduce(float v) {
  __shared__ float red[256];
  red[threadIdx.x] = v;
  __syncthreads();
  for (int s = 128; s > 0; s >>= 1) {
    if ((int)threadIdx.x < s) red[threadIdx.x] += red[threadIdx.x + s];
    __syncthreads();
  }
  float r = red[0];
  __syncthreads();
  return r;
}

// mean over n of a row pair; broadcast into f32 region + packed mirror (kOff)
__global__ __launch_bounds__(256) void pool_bcast_pack_kernel(
    const float* __restrict__ src, float* __restrict__ dstF,
    unsigned int* __restrict__ dstP, int kOff, int B3, int N, int cols)
{
  int seg = blockIdx.x;               // (C/2)*B3 blocks
  int cc = seg / B3, bd = seg % B3;
  int c0 = 2 * cc;
  size_t b0 = (size_t)c0 * cols + (size_t)bd * N, b1 = b0 + cols;
  float s0 = 0.0f, s1 = 0.0f;
  for (int n = threadIdx.x; n < N; n += 256) { s0 += src[b0 + n]; s1 += src[b1 + n]; }
  float m0 = block_reduce(s0) / (float)N;
  float m1 = block_reduce(s1) / (float)N;
  unsigned int pv = packbf2(m0, m1);
  for (int n = threadIdx.x; n < N; n += 256) {
    dstF[b0 + n] = m0; dstF[b1 + n] = m1;
    dstP[swz(kOff + c0, bd * N + n, cols)] = pv;
  }
}

__global__ __launch_bounds__(256) void pool_vec_kernel(
    const float* __restrict__ src, float* __restrict__ out,
    int B3, int N, int cols)
{
  int seg = blockIdx.x;
  int c = seg / B3, bd = seg % B3;
  size_t base = (size_t)c * cols + (size_t)bd * N;
  float s = 0.0f;
  for (int n = threadIdx.x; n < N; n += 256) s += src[base + n];
  float mean = block_reduce(s) / (float)N;
  if (threadIdx.x == 0) out[seg] = mean;
}

// broadcast g2[c, bd] over n into f32 region + packed mirror
__global__ void bcast_rows_pack_kernel(const float* __restrict__ vec,
                                       float* __restrict__ dstF,
                                       unsigned int* __restrict__ dstP,
                                       int kOff, int C2, int B3, int N)
{
  long tid = (long)blockIdx.x * blockDim.x + threadIdx.x;
  if (tid >= (long)C2 * B3 * N) return;
  int n  = (int)(tid % N);
  int bd = (int)((tid / N) % B3);
  int cc = (int)(tid / ((long)N * B3));
  int c0 = 2 * cc, cols = B3 * N;
  float v0 = vec[c0 * B3 + bd], v1 = vec[(c0 + 1) * B3 + bd];
  dstF[(size_t)c0 * cols + (size_t)bd * N + n] = v0;
  dstF[(size_t)(c0 + 1) * cols + (size_t)bd * N + n] = v1;
  dstP[swz(kOff + c0, bd * N + n, cols)] = packbf2(v0, v1);
}

// ---------------------------------------------------------------------------
// Stage kernels
// ---------------------------------------------------------------------------
__global__ __launch_bounds__(256) void center_kernel(
    const float* __restrict__ pcd, float* __restrict__ P,
    float* __restrict__ center)
{
  int row = blockIdx.x;                // b*3 + d
  size_t base = (size_t)row * N_;
  float s = 0.0f;
  for (int n = threadIdx.x; n < N_; n += 256) s += pcd[base + n];
  float mean = block_reduce(s) / (float)N_;
  for (int n = threadIdx.x; n < N_; n += 256) P[base + n] = pcd[base + n] - mean;
  if (threadIdx.x == 0) center[row] = mean;
}

__global__ __launch_bounds__(256) void knn_kernel(
    const float* __restrict__ P, int* __restrict__ idx)
{
  __shared__ float sx[256], sy[256], sz[256];
  int gq = blockIdx.x * 256 + threadIdx.x;
  int b = (blockIdx.x * 256) / N_;
  int n = gq - b * N_;
  size_t r0 = (size_t)(b * 3) * N_, r1 = r0 + N_, r2 = r1 + N_;
  float qx = P[r0 + n], qy = P[r1 + n], qz = P[r2 + n];
  float best[KNN]; int bidx[KNN];
  #pragma unroll
  for (int i = 0; i < KNN; ++i) { best[i] = -3.0e38f; bidx[i] = 0; }
  for (int t = 0; t < N_ / 256; ++t) {
    int m0 = t * 256;
    sx[threadIdx.x] = P[r0 + m0 + threadIdx.x];
    sy[threadIdx.x] = P[r1 + m0 + threadIdx.x];
    sz[threadIdx.x] = P[r2 + m0 + threadIdx.x];
    __syncthreads();
    for (int j = 0; j < 256; ++j) {
      float dx = qx - sx[j], dy = qy - sy[j], dz = qz - sz[j];
      float nd = -(dx*dx + dy*dy + dz*dz);
      if (nd > best[KNN - 1]) {
        int pos = KNN - 1;
        while (pos > 0 && best[pos - 1] < nd) {
          best[pos] = best[pos - 1]; bidx[pos] = bidx[pos - 1]; --pos;
        }
        best[pos] = nd; bidx[pos] = m0 + j;
      }
    }
    __syncthreads();
  }
  #pragma unroll
  for (int i = 0; i < KNN; ++i) idx[(size_t)gq * KNN + i] = bidx[i];
}

__global__ void graphfeat_kernel(const float* __restrict__ P,
                                 const int* __restrict__ idx,
                                 float* __restrict__ fbar)
{
  int gq = blockIdx.x * blockDim.x + threadIdx.x;
  if (gq >= B_ * N_) return;
  int b = gq / N_, n = gq % N_;
  size_t r0 = (size_t)(b * 3) * N_, r1 = r0 + N_, r2 = r1 + N_;
  float x0 = P[r0 + n], x1 = P[r1 + n], x2 = P[r2 + n];
  float sd0 = 0, sd1 = 0, sd2 = 0, sc0 = 0, sc1 = 0, sc2 = 0;
  for (int kk = 0; kk < KNN; ++kk) {
    int m = idx[(size_t)gq * KNN + kk];
    float a0 = P[r0 + m], a1 = P[r1 + m], a2 = P[r2 + m];
    sd0 += a0 - x0; sd1 += a1 - x1; sd2 += a2 - x2;
    sc0 += a1 * x2 - a2 * x1;
    sc1 += a2 * x0 - a0 * x2;
    sc2 += a0 * x1 - a1 * x0;
  }
  const float inv = 1.0f / (float)KNN;
  size_t c0 = (size_t)(b * 3 + 0) * N_ + n;
  size_t c1 = (size_t)(b * 3 + 1) * N_ + n;
  size_t c2 = (size_t)(b * 3 + 2) * N_ + n;
  fbar[0 * (size_t)COLS + c0] = sd0 * inv;
  fbar[0 * (size_t)COLS + c1] = sd1 * inv;
  fbar[0 * (size_t)COLS + c2] = sd2 * inv;
  fbar[1 * (size_t)COLS + c0] = x0;
  fbar[1 * (size_t)COLS + c1] = x1;
  fbar[1 * (size_t)COLS + c2] = x2;
  fbar[2 * (size_t)COLS + c0] = sc0 * inv;
  fbar[2 * (size_t)COLS + c1] = sc1 * inv;
  fbar[2 * (size_t)COLS + c2] = sc2 * inv;
}

__global__ void globalfeat_kernel(const float* __restrict__ g2,
                                  const float* __restrict__ center,
                                  float* __restrict__ out)
{
  int tid = blockIdx.x * blockDim.x + threadIdx.x;
  if (tid >= B_ * 128 * 3) return;
  int d = tid % 3, c = (tid / 3) % 128, b = tid / (3 * 128);
  out[tid] = g2[c * (B_ * 3) + b * 3 + d] + center[b * 3 + d];
}

// x_std invariants, packed-only output: row pairs of flat [192, B*N]
__global__ void xstd_pack_kernel(const float* __restrict__ lf,
                                 const float* __restrict__ z,
                                 unsigned int* __restrict__ flatp)
{
  long tid = (long)blockIdx.x * blockDim.x + threadIdx.x;
  if (tid >= (long)B_ * N_ * 96) return;
  int rp = (int)(tid % 96);
  int n  = (int)((tid / 96) % N_);
  int b  = (int)(tid / ((long)96 * N_));
  float v[2];
  #pragma unroll
  for (int p = 0; p < 2; ++p) {
    int r = 2 * rp + p, i = r / 3, kc = r % 3;
    size_t lbase = (size_t)i * COLS + (size_t)b * 3 * N_ + n;
    size_t zc = (size_t)b * 3 * N_ + (size_t)kc * N_ + n;
    v[p] = lf[lbase] * z[zc]
         + lf[lbase + N_] * z[(size_t)COLS + zc]
         + lf[lbase + 2 * (size_t)N_] * z[2 * (size_t)COLS + zc];
  }
  flatp[swz(2 * rp, b * N_ + n, COLSN)] = packbf2(v[0], v[1]);
}

__global__ __launch_bounds__(256) void bn_relu_kernel(
    const float* __restrict__ y, const float* __restrict__ gw,
    const float* __restrict__ bw, float* __restrict__ outLocal)
{
  int o = blockIdx.x;
  size_t base = (size_t)o * COLSN;
  float s = 0.0f, sq = 0.0f;
  for (int c = threadIdx.x; c < COLSN; c += 256) {
    float v = y[base + c]; s += v; sq += v * v;
  }
  float sum = block_reduce(s);
  float ssq = block_reduce(sq);
  float mean = sum / (float)COLSN;
  float var = ssq / (float)COLSN - mean * mean;
  float inv = rsqrtf(var + 1e-5f);
  float gg = gw[o], bb = bw[o];
  for (int c = threadIdx.x; c < COLSN; c += 256) {
    int b = c / N_, n = c % N_;
    float v = (y[base + c] - mean) * inv * gg + bb;
    outLocal[(size_t)b * 64 * N_ + (size_t)o * N_ + n] = v > 0.0f ? v : 0.0f;
  }
}

// ---------------------------------------------------------------------------
// Host orchestration
// ---------------------------------------------------------------------------
static inline void launch_wmma(hipStream_t s, const unsigned int* Wp,
                               const unsigned int* Xp, const float* Add,
                               float* Yf, unsigned int* Yp, int O, int I, int M) {
  int waves = (O / 16) * (M / 64);
  hipLaunchKernelGGL(wmma_gemm_kernel, dim3((waves + 3) / 4), dim3(128), 0, s,
                     Wp, Xp, Add, Yf, Yp, O, I, M);
}
static inline void launch_lrelu(hipStream_t s, const float* x, const float* d,
                                float* outF, unsigned int* outP,
                                int C, int N, int Bv, float slope) {
  long total = (long)Bv * (C / 2) * N;
  hipLaunchKernelGGL(vn_lrelu_pack_kernel, dim3((unsigned)((total + 255) / 256)),
                     dim3(256), 0, s, x, d, outF, outP, C / 2, N, Bv, slope);
}

static inline float* bump(char*& p, size_t nwords) {   // 4-byte words
  float* r = (float*)p;
  p += ((nwords * 4 + 255) & ~(size_t)255);
  return r;
}

extern "C" void kernel_launch(void* const* d_in, const int* in_sizes, int n_in,
                              void* d_out, int out_size, void* d_ws, size_t ws_size,
                              hipStream_t stream) {
  (void)n_in; (void)out_size; (void)ws_size;
  auto f = [&](int i) { return (const float*)d_in[i]; };

  const float *pcd, *W_pre, *d_glob, *W_glob, *std_W1, *std_d1, *std_W2,
              *std_d2, *std_Wlin, *W_loc, *bn_g, *bn_b;
  const float* blk[7][5];   // [block][d0,W0,d1,W1,Ws]

  if (in_sizes[0] == 1) {   // jax pytree (sorted-keys) flattening
    W_glob = f(1); W_loc = f(2); W_pre = f(3); bn_b = f(4); bn_g = f(5);
    d_glob = f(6);
    int i = 7;
    for (int bl = 0; bl < 7; ++bl) {   // W0, W1, Ws, d0, d1
      const float* W0 = f(i++); const float* W1 = f(i++);
      const float* Ws = f(i++); const float* d0 = f(i++); const float* d1 = f(i++);
      blk[bl][0] = d0; blk[bl][1] = W0; blk[bl][2] = d1;
      blk[bl][3] = W1; blk[bl][4] = Ws;
    }
    std_W1 = f(42); std_W2 = f(43); std_Wlin = f(44);
    std_d1 = f(45); std_d2 = f(46); pcd = f(47);
  } else {                  // dict insertion-order flattening
    pcd = f(0); W_pre = f(2);
    int i = 3;
    for (int bl = 0; bl < 5; ++bl) {
      blk[bl][0] = f(i++); blk[bl][1] = f(i++); blk[bl][2] = f(i++);
      blk[bl][3] = f(i++); blk[bl][4] = f(i++);
    }
    d_glob = f(28); W_glob = f(29);
    i = 30;
    for (int bl = 5; bl < 7; ++bl) {
      blk[bl][0] = f(i++); blk[bl][1] = f(i++); blk[bl][2] = f(i++);
      blk[bl][3] = f(i++); blk[bl][4] = f(i++);
    }
    std_W1 = f(40); std_d1 = f(41); std_W2 = f(42); std_d2 = f(43);
    std_Wlin = f(44); W_loc = f(45); bn_g = f(46); bn_b = f(47);
  }

  const size_t BIG = (size_t)256 * COLS, HALF = (size_t)128 * COLS;
  char* wp = (char*)d_ws;
  float* P      = bump(wp, B_ * 3 * N_);
  float* center = bump(wp, 8);
  int*   idx    = (int*)bump(wp, (size_t)B_ * N_ * KNN);
  float* fbar   = bump(wp, 3 * (size_t)COLS);
  float* bufA   = bump(wp, BIG);
  float* bufB   = bump(wp, BIG);
  float* T1     = bump(wp, BIG);
  float* T2     = bump(wp, HALF);
  float* T3     = bump(wp, HALF);
  float* latent = bump(wp, HALF);
  float* gbuf   = bump(wp, 128 * 6);
  float* dG     = bump(wp, 128 * 6);
  float* g2     = bump(wp, 128 * 6);
  float* t32    = bump(wp, (size_t)32 * COLS);
  float* d32    = bump(wp, (size_t)32 * COLS);
  float* t16    = bump(wp, (size_t)16 * COLS);
  float* d16    = bump(wp, (size_t)16 * COLS);
  float* zbuf   = bump(wp, (size_t)3 * COLS);
  float* ybuf   = bump(wp, (size_t)64 * COLSN);
  // packed (u32) mirrors
  unsigned int* bufAp   = (unsigned int*)bump(wp, (size_t)128 * COLS);
  unsigned int* bufBp   = (unsigned int*)bump(wp, (size_t)128 * COLS);
  unsigned int* T1p     = (unsigned int*)bump(wp, (size_t)128 * COLS);
  unsigned int* T2p     = (unsigned int*)bump(wp, (size_t)64 * COLS);
  unsigned int* T3p     = (unsigned int*)bump(wp, (size_t)64 * COLS);
  unsigned int* latentp = (unsigned int*)bump(wp, (size_t)64 * COLS);
  unsigned int* t32p    = (unsigned int*)bump(wp, (size_t)16 * COLS);
  unsigned int* a32p    = (unsigned int*)bump(wp, (size_t)16 * COLS);
  unsigned int* flatp   = (unsigned int*)bump(wp, (size_t)96 * COLSN);

  // weight arena + packers
  auto packW = [&](const float* Wf, int O, int I) -> const unsigned int* {
    long total = (long)O * I / 2;
    unsigned int* dst = (unsigned int*)bump(wp, (size_t)total);
    hipLaunchKernelGGL(pack_weight_kernel, dim3((unsigned)((total + 255) / 256)),
                       dim3(256), 0, stream, Wf, dst, total);
    return dst;
  };
  const int bcin[7] = {256,256,256,256,256,256,128};
  const int bh[7]   = {128,128,128,128,128, 64, 64};
  const int bco[7]  = {128,128,128,128,128, 64, 64};
  const unsigned int* pblk[7][5];
  for (int bl = 0; bl < 7; ++bl) {
    pblk[bl][0] = packW(blk[bl][0], bcin[bl], bcin[bl]);   // d0
    pblk[bl][1] = packW(blk[bl][1], bh[bl],   bcin[bl]);   // W0
    pblk[bl][2] = packW(blk[bl][2], bh[bl],   bh[bl]);     // d1
    pblk[bl][3] = packW(blk[bl][3], bco[bl],  bh[bl]);     // W1
    pblk[bl][4] = packW(blk[bl][4], bco[bl],  bcin[bl]);   // Ws
  }
  const unsigned int* pW1   = packW(std_W1, 32, 64);
  const unsigned int* pd1s  = packW(std_d1, 32, 32);
  const unsigned int* pW2   = packW(std_W2, 16, 32);
  const unsigned int* pWloc = packW(W_loc, 64, 192);

  float* outGlobal = (float*)d_out;
  float* outLocal  = (float*)d_out + B_ * 128 * 3;
  const int B3 = B_ * 3;

  // full VN resnet block (f32 + packed dataflow)
  auto run_block = [&](const float* xF, const unsigned int* xP, int bl,
                       float* outF, unsigned int* outP) {
    int cin = bcin[bl], h = bh[bl], cout = bco[bl];
    launch_wmma(stream, pblk[bl][0], xP, nullptr, T1, nullptr, cin, cin, COLS);
    launch_lrelu(stream, xF, T1, nullptr, T1p, cin, N_, B_, 0.0f);        // a0
    launch_wmma(stream, pblk[bl][1], T1p, nullptr, T2, T2p, h, cin, COLS);
    launch_wmma(stream, pblk[bl][2], T2p, nullptr, T3, nullptr, h, h, COLS);
    launch_lrelu(stream, T2, T3, nullptr, T3p, h, N_, B_, 0.0f);          // a1
    launch_wmma(stream, pblk[bl][4], xP, nullptr, outF, nullptr, cout, cin, COLS);
    launch_wmma(stream, pblk[bl][3], T3p, outF, outF, outP, cout, h, COLS);
  };

  // 1-3) center, kNN, graph-feature means
  hipLaunchKernelGGL(center_kernel, dim3(B3), dim3(256), 0, stream, pcd, P, center);
  hipLaunchKernelGGL(knn_kernel, dim3(B_ * N_ / 256), dim3(256), 0, stream, P, idx);
  hipLaunchKernelGGL(graphfeat_kernel, dim3((B_ * N_ + 255) / 256), dim3(256), 0,
                     stream, P, idx, fbar);

  // 4) feat0 = W_pre @ fbar (K=3) -> bufA f32 + packed
  {
    long total = (long)128 * COLS;
    hipLaunchKernelGGL(gemm_naive_pack_kernel, dim3((unsigned)((total + 255) / 256)),
                       dim3(256), 0, stream, W_pre, fbar, bufA, bufAp, 256, 3, COLS);
  }

  // 5-9) net0
  run_block(bufA, bufAp, 0, bufB, bufBp);
  hipLaunchKernelGGL(pool_bcast_pack_kernel, dim3(64 * B3), dim3(256), 0, stream,
                     bufB, bufB + HALF, bufBp, 128, B3, N_, COLS);
  run_block(bufB, bufBp, 1, bufA, bufAp);
  hipLaunchKernelGGL(pool_bcast_pack_kernel, dim3(64 * B3), dim3(256), 0, stream,
                     bufA, bufA + HALF, bufAp, 128, B3, N_, COLS);
  run_block(bufA, bufAp, 2, latent, latentp);

  // 10-12) net1 on concat(latent, pool(latent))
  hipMemcpyAsync(bufB, latent, HALF * sizeof(float), hipMemcpyDeviceToDevice, stream);
  hipMemcpyAsync(bufBp, latentp, (size_t)64 * COLS * 4, hipMemcpyDeviceToDevice, stream);
  hipLaunchKernelGGL(pool_bcast_pack_kernel, dim3(64 * B3), dim3(256), 0, stream,
                     latent, bufB + HALF, bufBp, 128, B3, N_, COLS);
  run_block(bufB, bufBp, 3, bufA, bufAp);
  hipLaunchKernelGGL(pool_bcast_pack_kernel, dim3(64 * B3), dim3(256), 0, stream,
                     bufA, bufA + HALF, bufAp, 128, B3, N_, COLS);
  run_block(bufA, bufAp, 4, bufB, bufBp);     // gnet -> bufB[128 rows]

  // 13-15) global branch (M = 6, f32 naive)
  hipLaunchKernelGGL(pool_vec_kernel, dim3(128 * B3), dim3(256), 0, stream,
                     bufB, gbuf, B3, N_, COLS);
  hipLaunchKernelGGL(gemm_naive_kernel, dim3(3), dim3(256), 0, stream,
                     d_glob, gbuf, dG, 128, 128, 6);
  launch_lrelu(stream, gbuf, dG, dG, nullptr, 128, 1, B_, 0.2f);
  hipLaunchKernelGGL(gemm_naive_kernel, dim3(3), dim3(256), 0, stream,
                     W_glob, dG, g2, 128, 128, 6);
  hipLaunchKernelGGL(globalfeat_kernel, dim3(3), dim3(256), 0, stream,
                     g2, center, outGlobal);

  // 16-18) net2 on concat(latent, broadcast(g2))
  hipMemcpyAsync(bufB, latent, HALF * sizeof(float), hipMemcpyDeviceToDevice, stream);
  hipMemcpyAsync(bufBp, latentp, (size_t)64 * COLS * 4, hipMemcpyDeviceToDevice, stream);
  {
    long total = (long)64 * B3 * N_;
    hipLaunchKernelGGL(bcast_rows_pack_kernel, dim3((unsigned)((total + 255) / 256)),
                       dim3(256), 0, stream, g2, bufB + HALF, bufBp, 128, 64, B3, N_);
  }
  run_block(bufB, bufBp, 5, bufA, bufAp);
  hipLaunchKernelGGL(pool_bcast_pack_kernel, dim3(32 * B3), dim3(256), 0, stream,
                     bufA, bufA + (size_t)64 * COLS, bufAp, 64, B3, N_, COLS);
  run_block(bufA, bufAp, 6, latent, latentp);  // lf -> latent (64 rows)

  // 19) std branch
  launch_wmma(stream, pW1, latentp, nullptr, t32, t32p, 32, 64, COLS);
  launch_wmma(stream, pd1s, t32p, nullptr, d32, nullptr, 32, 32, COLS);
  launch_lrelu(stream, t32, d32, nullptr, a32p, 32, N_, B_, 0.2f);
  launch_wmma(stream, pW2, a32p, nullptr, t16, nullptr, 16, 32, COLS);
  {
    long total = (long)16 * COLS;
    hipLaunchKernelGGL(gemm_naive_kernel, dim3((unsigned)((total + 255) / 256)),
                       dim3(256), 0, stream, std_d2, t16, d16, 16, 16, COLS);
  }
  launch_lrelu(stream, t16, d16, d16, nullptr, 16, N_, B_, 0.2f);
  {
    long total = (long)3 * COLS;
    hipLaunchKernelGGL(gemm_naive_kernel, dim3((unsigned)((total + 255) / 256)),
                       dim3(256), 0, stream, std_Wlin, d16, zbuf, 3, 16, COLS);
  }

  // 20-22) invariants, W_loc GEMM, batchnorm + relu
  {
    long total = (long)B_ * N_ * 96;
    hipLaunchKernelGGL(xstd_pack_kernel, dim3((unsigned)((total + 255) / 256)),
                       dim3(256), 0, stream, latent, zbuf, flatp);
  }
  launch_wmma(stream, pWloc, flatp, nullptr, ybuf, nullptr, 64, 192, COLSN);
  hipLaunchKernelGGL(bn_relu_kernel, dim3(64), dim3(256), 0, stream,
                     ybuf, bn_g, bn_b, outLocal);
}
